// LinearAttentionMem_78683800863377
// MI455X (gfx1250) — matrix operand order repaired and checked by reading, not compiled, not run on hardware
//
#include <hip/hip_runtime.h>

// LinearAttentionMem fused kernel for gfx1250 (MI455X).
// Chunked closed-form of the f-gated rank-1 scan:
//   per chunk (C=128 steps):  M = (prod f) * M + Ktilde^T @ V
//   Ktilde[t,:] = suffixprod(f)[t+1..] * w[t] * k[t,:]
// Three 128x128x128 GEMMs per chunk on v_wmma_f32_16x16x32_f16 (96/chunk).
// One workgroup (8 x wave32) per (b,h). State (128x128 f32) lives in VGPRs.
// GEMM operands staged in LDS; next chunk's x streams in concurrently via
// GLOBAL_LOAD_ASYNC_TO_LDS_B128 (ASYNCcnt) into a double buffer.

#define B_  16
#define S_  2048
#define D_  128
#define H_  8
#define NM_ 4
#define C_  128   // timesteps per chunk (S_/C_ = 16 chunks)

typedef __attribute__((ext_vector_type(16))) _Float16 v16h;
typedef __attribute__((ext_vector_type(8)))  _Float16 v8h;
typedef __attribute__((ext_vector_type(4)))  _Float16 v4h;
typedef __attribute__((ext_vector_type(8)))  float    v8f;

union FragA { v16h v; v8h h[2]; };

__device__ __forceinline__ void async_load16B(unsigned lds_byte_off,
                                              const void* gaddr) {
  unsigned long long ga = (unsigned long long)(uintptr_t)gaddr;
  asm volatile("global_load_async_to_lds_b128 %0, %1, off"
               :: "v"(lds_byte_off), "v"(ga) : "memory");
}

__global__ __launch_bounds__(256, 1)
void lam_fused(const float* __restrict__ x,
               const float* __restrict__ M0,
               const float* __restrict__ router,
               const int*   __restrict__ mem_id,
               const float* __restrict__ Wk,
               const float* __restrict__ bk,
               const float* __restrict__ Wv,
               const float* __restrict__ bv,
               const float* __restrict__ wf,
               const float* __restrict__ bf,
               float* __restrict__ out)
{
  __shared__ __align__(16) _Float16 wk_lds[D_ * D_];  // Wk[:, h*D:(h+1)*D] f16
  __shared__ __align__(16) _Float16 wv_lds[D_ * D_];  // Wv slice, f16
  __shared__ __align__(16) _Float16 x_lds [C_ * D_];  // x chunk   [t][d] f16
  __shared__ __align__(16) _Float16 kT_lds[D_ * C_];  // Ktilde^T  [d][t]
  __shared__ __align__(16) _Float16 v_lds [C_ * D_];  // V         [t][d]
  __shared__ __align__(16) float    xs32[2][C_ * D_]; // async f32 double buffer
  __shared__ float scan_a[C_];
  __shared__ float scan_b[C_];
  __shared__ float coeff [C_];
  __shared__ float wf_col[D_];
  __shared__ float bk_l  [D_];
  __shared__ float bv_l  [D_];
  __shared__ float Pchunk;

  const int tid   = threadIdx.x;
  const int wv_   = tid >> 5;        // wave id 0..7 (wave32)
  const int lane  = tid & 31;
  const int ln15  = lane & 15;
  const int hi    = (lane >> 4) & 1;
  const int kbase = hi * 8;          // A-frag K-base (ISA 7.12.2 layout)
  const int mbase = 16 * wv_ + hi * 8;

  const int bh  = blockIdx.x;
  const int b   = bh / H_;
  const int h   = bh % H_;
  const int mid = mem_id[0];
  const float bf_h = bf[h];
  const float* xg = x + (size_t)b * S_ * D_;

  // ---- kick off async load of chunk 0 into buffer 0 (overlaps prologue) ----
  {
    #pragma unroll
    for (int q = 0; q < 16; ++q) {
      const int idx = q * 256 + tid;        // 0..4095 float4 slots
      const unsigned lo = (unsigned)(uintptr_t)(&xs32[0][idx * 4]);
      async_load16B(lo, xg + (size_t)idx * 4);
    }
  }

  // ---- stage per-head constants & weight slices (once) ----
  if (tid < D_) {
    wf_col[tid] = wf[tid * H_ + h];
    bk_l[tid]   = bk[h * D_ + tid];
    bv_l[tid]   = bv[h * D_ + tid];
  }
  {
    const float* wkg = Wk + h * D_;
    const float* wvg = Wv + h * D_;
    #pragma unroll
    for (int q = 0; q < 16; ++q) {
      const int idx = q * 256 + tid;        // 0..4095 float4 slots
      const int row = idx >> 5;             // 0..127 (input dim)
      const int c4  = (idx & 31) << 2;      // 0..124 (head col, step 4)
      const float4 kx = *(const float4*)(wkg + (size_t)row * (D_ * H_) + c4);
      const float4 vx = *(const float4*)(wvg + (size_t)row * (D_ * H_) + c4);
      v4h kh = { (_Float16)kx.x, (_Float16)kx.y, (_Float16)kx.z, (_Float16)kx.w };
      v4h vh = { (_Float16)vx.x, (_Float16)vx.y, (_Float16)vx.z, (_Float16)vx.w };
      *(v4h*)&wk_lds[row * D_ + c4] = kh;
      *(v4h*)&wv_lds[row * D_ + c4] = vh;
    }
  }

  // ---- load initial state: wave w owns M rows [16w,16w+16), 8 col tiles ----
  v8f Macc[8];
  {
    const float* Mg = M0 + (size_t)bh * D_ * D_;
    #pragma unroll
    for (int j = 0; j < 8; ++j) {
      const int dn = 16 * j + ln15;
      #pragma unroll
      for (int r = 0; r < 8; ++r)
        Macc[j][r] = Mg[(mbase + r) * D_ + dn];
    }
  }

  int cur = 0;
  for (int c0 = 0; c0 < S_; c0 += C_) {
    // ---- wait for this chunk's async x tile, then prefetch the next one ----
    asm volatile("s_wait_asynccnt 0x0" ::: "memory");
    __syncthreads();
    if (c0 + C_ < S_) {
      const float* xn = xg + (size_t)(c0 + C_) * D_;
      #pragma unroll
      for (int q = 0; q < 16; ++q) {
        const int idx = q * 256 + tid;
        const unsigned lo = (unsigned)(uintptr_t)(&xs32[cur ^ 1][idx * 4]);
        async_load16B(lo, xn + (size_t)idx * 4);
      }
    }
    // ---- convert staged f32 -> f16 x_lds ----
    {
      #pragma unroll
      for (int q = 0; q < 16; ++q) {
        const int idx = q * 256 + tid;
        const float4 fx = *(const float4*)&xs32[cur][idx * 4];
        v4h hx = { (_Float16)fx.x, (_Float16)fx.y, (_Float16)fx.z, (_Float16)fx.w };
        *(v4h*)&x_lds[idx * 4] = hx;
      }
    }
    __syncthreads();

    // ---- phase 1: gates f = sigmoid(x@wf+bf), router w, suffix products ----
    float w_reg = 0.0f;
    if (tid < C_) {
      const _Float16* xr = &x_lds[tid * D_];
      float acc = bf_h;
      #pragma unroll 8
      for (int d = 0; d < D_; ++d) acc += (float)xr[d] * wf_col[d];
      scan_a[tid] = 1.0f / (1.0f + __expf(-acc));
      const size_t ridx = ((size_t)b * S_ + c0 + tid) * NM_ + mid;
      w_reg = router[ridx];
      if (c0 + C_ < S_)
        __builtin_prefetch(&router[ridx + (size_t)C_ * NM_], 0, 1);
    }
    __syncthreads();
    {   // Hillis-Steele suffix-inclusive product: S[t] = prod_{t'>=t} f[t']
      float* srcp = scan_a; float* dstp = scan_b;
      #pragma unroll
      for (int off = 1; off < C_; off <<= 1) {
        if (tid < C_) {
          float tail = (tid + off < C_) ? srcp[tid + off] : 1.0f;
          dstp[tid] = srcp[tid] * tail;
        }
        __syncthreads();
        float* t_ = srcp; srcp = dstp; dstp = t_;
      }
      if (tid < C_) coeff[tid] = ((tid + 1 < C_) ? srcp[tid + 1] : 1.0f) * w_reg;
      if (tid == 0) Pchunk = srcp[0];
    }
    __syncthreads();

    // per-wave coefficient registers for this chunk's time rows
    float cf[8];
    #pragma unroll
    for (int r = 0; r < 8; ++r) cf[r] = coeff[mbase + r];

    // ---- phase 2: K = x@Wk+bk, V = x@Wv+bv (WMMA); store scaled K^T, V ----
    #pragma unroll
    for (int j = 0; j < 8; ++j) {
      v8f acck = {};
      v8f accv = {};
      #pragma unroll
      for (int s = 0; s < 4; ++s) {
        FragA a;   // x rows 16w..16w+15 (time), K-cols 32s..32s+31
        const _Float16* xp = &x_lds[(16 * wv_ + ln15) * D_ + 32 * s + kbase];
        a.h[0] = *(const v8h*)(xp);
        a.h[1] = *(const v8h*)(xp + 16);
        v16h bkf;
        for (int i = 0; i < 16; ++i) bkf[i] = wk_lds[(32 * s + kbase + (i & 7) + ((i >> 3) << 4)) * D_ + 16 * j + ln15];
        acck = __builtin_amdgcn_wmma_f32_16x16x32_f16(
            false, a.v, false, bkf, (short)0, acck, false, false);
        v16h bvf;
        for (int i = 0; i < 16; ++i) bvf[i] = wv_lds[(32 * s + kbase + (i & 7) + ((i >> 3) << 4)) * D_ + 16 * j + ln15];
        accv = __builtin_amdgcn_wmma_f32_16x16x32_f16(
            false, a.v, false, bvf, (short)0, accv, false, false);
      }
      const int dn  = 16 * j + ln15;
      const float bkv = bk_l[dn];
      const float bvv = bv_l[dn];
      v8h kp;
      #pragma unroll
      for (int r = 0; r < 8; ++r) {
        kp[r] = (_Float16)((acck[r] + bkv) * cf[r]);
        v_lds[(mbase + r) * D_ + dn] = (_Float16)(accv[r] + bvv);
      }
      *(v8h*)&kT_lds[dn * C_ + mbase] = kp;   // contiguous 8 rows -> one b128
    }
    __syncthreads();

    // ---- phase 3: M = P*M + Ktilde^T @ V ----
    const float P = Pchunk;
    #pragma unroll
    for (int j = 0; j < 8; ++j) {
      v8f m_ = Macc[j];
      #pragma unroll
      for (int r = 0; r < 8; ++r) m_[r] *= P;
      #pragma unroll
      for (int s = 0; s < 4; ++s) {
        FragA a;   // Ktilde^T rows 16w..16w+15 (d), K-dim = time
        const _Float16* kp2 = &kT_lds[(16 * wv_ + ln15) * C_ + 32 * s + kbase];
        a.h[0] = *(const v8h*)(kp2);
        a.h[1] = *(const v8h*)(kp2 + 16);
        v16h bfr;
        for (int i = 0; i < 16; ++i) bfr[i] = v_lds[(32 * s + kbase + (i & 7) + ((i >> 3) << 4)) * D_ + 16 * j + ln15];
        m_ = __builtin_amdgcn_wmma_f32_16x16x32_f16(
            false, a.v, false, bfr, (short)0, m_, false, false);
      }
      Macc[j] = m_;
    }
    __syncthreads();
    cur ^= 1;
  }

  // ---- write final state ----
  {
    float* Og = out + (size_t)bh * D_ * D_;
    #pragma unroll
    for (int j = 0; j < 8; ++j) {
      const int dn = 16 * j + ln15;
      #pragma unroll
      for (int r = 0; r < 8; ++r)
        Og[(mbase + r) * D_ + dn] = Macc[j][r];
    }
  }
}

extern "C" void kernel_launch(void* const* d_in, const int* in_sizes, int n_in,
                              void* d_out, int out_size, void* d_ws, size_t ws_size,
                              hipStream_t stream) {
  const float* x      = (const float*)d_in[0];
  const float* M0     = (const float*)d_in[1];
  const float* router = (const float*)d_in[2];
  const int*   mem_id = (const int*)  d_in[3];
  const float* Wk     = (const float*)d_in[4];
  const float* bk     = (const float*)d_in[5];
  const float* Wv     = (const float*)d_in[6];
  const float* bv     = (const float*)d_in[7];
  const float* Wf     = (const float*)d_in[8];
  const float* bf     = (const float*)d_in[9];
  float* out = (float*)d_out;
  (void)d_ws; (void)ws_size; (void)in_sizes; (void)n_in; (void)out_size;

  lam_fused<<<B_ * H_, 256, 0, stream>>>(x, M0, router, mem_id,
                                         Wk, bk, Wv, bv, Wf, bf, out);
}
